// SparseMOE_62156766707958
// MI455X (gfx1250) — matrix operand order repaired and checked by reading, not compiled, not run on hardware
//
#include <hip/hip_runtime.h>
#include <hip/hip_bf16.h>
#include <math.h>

// Problem constants (match reference)
constexpr int BB = 4, SS = 2048, HH = 1024, EE = 8;
constexpr int NN = BB * SS;          // 8192 tokens
constexpr int DFF = HH * 8 / 3;      // 2730
constexpr int NKROWS = NN * 2;       // 16384 routed (token,expert) rows
constexpr int UPLD = 2816;           // DFF padded to 22*128 (zero-filled)
constexpr int MAXRT = NKROWS / 128;  // 128 row-tiles per expert worst case

typedef __attribute__((ext_vector_type(16))) __bf16 v16bf;
typedef __attribute__((ext_vector_type(8)))  float  v8f;

union Frag { v16bf v; uint4 q[2]; };

__device__ __forceinline__ unsigned short f2bf(float f) {
  unsigned int u = __float_as_uint(f);
  u += 0x7FFFu + ((u >> 16) & 1u);           // round-to-nearest-even
  return (unsigned short)(u >> 16);
}

// ---------------------------------------------------------------- cast f32->bf16
__global__ __launch_bounds__(256) void cast_bf16_kernel(
    const float* __restrict__ s, unsigned short* __restrict__ d, long n) {
  long i = (long)blockIdx.x * 256 + threadIdx.x;
  if (i < n) d[i] = f2bf(s[i]);
}

// ------------------------------------------------- cast + transpose weights
// W[e][k][n] (f32) -> WT[e][n][kpad] (bf16), zero-filled for k >= Kdim.
// K-contiguous rows make the B tile a straight async 16B copy into LDS.
__global__ __launch_bounds__(256) void cast_transpose_kernel(
    const float* __restrict__ W, unsigned short* __restrict__ WT,
    int Kdim, int Ncols, int kpad, long total) {
  long i = (long)blockIdx.x * 256 + threadIdx.x;
  if (i >= total) return;
  int k = (int)(i % kpad);
  long t = i / kpad;
  int n = (int)(t % Ncols);
  int e = (int)(t / Ncols);
  float v = (k < Kdim) ? W[((size_t)e * Kdim + k) * Ncols + n] : 0.f;
  WT[i] = f2bf(v);
}

// ---------------------------------------------------------------- router
// one wave (32 lanes) per token: 8 logits, softmax-free top-2 renorm weights
__global__ __launch_bounds__(256) void router_kernel(
    const float* __restrict__ x, const float* __restrict__ Wr,
    const float* __restrict__ br, float* __restrict__ ew,
    int* __restrict__ topE, float* __restrict__ topW, int* __restrict__ cnt) {
  int wid = threadIdx.x >> 5, lane = threadIdx.x & 31;
  int t = blockIdx.x * 8 + wid;
  float a[EE];
#pragma unroll
  for (int e = 0; e < EE; ++e) a[e] = 0.f;
  const float* xr = x + (size_t)t * HH;
  for (int h = lane; h < HH; h += 32) {
    float xv = xr[h];
    const float* wr = Wr + (size_t)h * EE;
#pragma unroll
    for (int e = 0; e < EE; ++e) a[e] = fmaf(xv, wr[e], a[e]);
  }
#pragma unroll
  for (int e = 0; e < EE; ++e)
    for (int off = 16; off; off >>= 1) a[e] += __shfl_xor(a[e], off, 32);
  if (lane == 0) {
    float l[EE];
#pragma unroll
    for (int e = 0; e < EE; ++e) { l[e] = a[e] + br[e]; ew[(size_t)t * EE + e] = l[e]; }
    int i0 = 0; float m0 = l[0];
#pragma unroll
    for (int e = 1; e < EE; ++e) if (l[e] > m0) { m0 = l[e]; i0 = e; }
    int i1 = (i0 == 0) ? 1 : 0; float m1 = l[i1];
#pragma unroll
    for (int e = 0; e < EE; ++e)
      if (e != i0 && l[e] > m1) { m1 = l[e]; i1 = e; }
    // softmax + top2 renorm collapses to pairwise sigmoid of logit gap
    float w0 = 1.f / (1.f + __expf(m1 - m0));
    topE[2 * t] = i0; topE[2 * t + 1] = i1;
    topW[2 * t] = w0; topW[2 * t + 1] = 1.f - w0;
    atomicAdd(&cnt[i0], 1);
    atomicAdd(&cnt[i1], 1);
  }
}

// ---------------------------------------------------------------- scan (E=8)
__global__ void scan_kernel(const int* __restrict__ cnt, int* __restrict__ off) {
  int s = 0;
  for (int e = 0; e < EE; ++e) { off[e] = s; s += cnt[e]; }
}

// ---------------------------------------------------------------- scatter tokens
__global__ __launch_bounds__(256) void scatter_kernel(
    const int* __restrict__ topE, const float* __restrict__ topW,
    const int* __restrict__ off, int* __restrict__ cur,
    int* __restrict__ tokRow, float* __restrict__ rowW) {
  int t = blockIdx.x * 256 + threadIdx.x;
  if (t >= NN) return;
#pragma unroll
  for (int k = 0; k < 2; ++k) {
    int e = topE[2 * t + k];
    int pos = atomicAdd(&cur[e], 1);
    int slot = off[e] + pos;
    tokRow[slot] = t;
    rowW[slot] = topW[2 * t + k];
  }
}

// ---------------------------------------------------------------- grouped GEMM
// Double-buffered LDS tiles, staged with GLOBAL_LOAD_ASYNC_TO_LDS_B128:
// iteration k issues tile k+1 into the ping-pong buffer, waits asynccnt<=2
// (in-order completion => tile k landed), then runs 8 WMMAs while the async
// engine streams tile k+1. EXEC stays all-ones via address clamping.
// MODE 0: gate  — A rows gathered via tokRow, out bf16
// MODE 1: up    — A rows contiguous,      out bf16 with SiLU, zero-padded cols
// MODE 2: down  — A rows contiguous, f32 atomic scatter-add (x combine weight)
template <int MODE>
__global__ __launch_bounds__(256) void moe_gemm(
    const unsigned short* __restrict__ A, int lda,
    const unsigned short* __restrict__ BT, int ldbt,
    int Ncols, int Kloop, size_t strideB,
    const float* __restrict__ bias,
    unsigned short* __restrict__ Out, int ldo,
    float* __restrict__ Y,
    const int* __restrict__ segOff, const int* __restrict__ segCnt,
    const int* __restrict__ tokRow, const float* __restrict__ rowW) {
  __shared__ __align__(16) unsigned short As[2][128][40];  // ping-pong 128x32 (pad 40)
  __shared__ __align__(16) unsigned short Bs[2][128][40];

  const int e  = blockIdx.y >> 7;        // / MAXRT
  const int rt = blockIdx.y & (MAXRT - 1);
  const int rows = segCnt[e];
  const int r0 = rt * 128;
  if (r0 >= rows) return;
  const int n0 = blockIdx.x * 128;
  const int seg0 = segOff[e];

  const int tid = threadIdx.x;
  const int wid = tid >> 5, lane = tid & 31;
  const int waveM = wid >> 1, waveN = wid & 1;  // 4x2 wave grid -> 32x64 per wave
  const int sRow = tid >> 1, sHalf = tid & 1;   // staging: row/col, 16-elem half

  // Loop-invariant staging addresses. Out-of-range rows/cols are CLAMPED, not
  // predicated: garbage rows feed only output rows the epilogue masks, and
  // keeping EXEC all-ones is what the async + WMMA path wants.
  long arow;
  {
    int rl = r0 + sRow;
    if (MODE == 0) arow = (rl < rows) ? (long)tokRow[seg0 + rl] : 0L;
    else           arow = (rl < rows) ? (long)(seg0 + rl) : (long)seg0;
  }
  const unsigned short* aSrc = A + (size_t)arow * lda + sHalf * 16;
  int bcol = n0 + sRow; if (bcol >= Ncols) bcol = Ncols - 1;
  const unsigned short* bSrc = BT + (size_t)e * strideB + (size_t)bcol * ldbt + sHalf * 16;
  const unsigned ldsA0 = (unsigned)(uintptr_t)&As[0][sRow][sHalf * 16];
  const unsigned ldsA1 = (unsigned)(uintptr_t)&As[1][sRow][sHalf * 16];
  const unsigned ldsB0 = (unsigned)(uintptr_t)&Bs[0][sRow][sHalf * 16];
  const unsigned ldsB1 = (unsigned)(uintptr_t)&Bs[1][sRow][sHalf * 16];

  v8f acc[2][4];
#pragma unroll
  for (int mi = 0; mi < 2; ++mi)
#pragma unroll
    for (int ni = 0; ni < 4; ++ni) acc[mi][ni] = (v8f)0.f;

  const int T = Kloop >> 5;  // K-tiles of 32
  // prologue: tile 0 -> buffer 0
  asm volatile("global_load_async_to_lds_b128 %0, %1, off"
               :: "v"(ldsA0), "v"(aSrc) : "memory");
  asm volatile("global_load_async_to_lds_b128 %0, %1, off"
               :: "v"(ldsB0), "v"(bSrc) : "memory");

  for (int kt = 0; kt < T; ++kt) {
    const int p = kt & 1;
    if (kt + 1 < T) {  // issue tile kt+1 into the other buffer (uniform branch)
      const unsigned short* an = aSrc + (kt + 1) * 32;
      const unsigned short* bn = bSrc + (kt + 1) * 32;
      asm volatile("global_load_async_to_lds_b128 %0, %1, off"
                   :: "v"(p ? ldsA0 : ldsA1), "v"(an) : "memory");
      asm volatile("global_load_async_to_lds_b128 %0, %1, off"
                   :: "v"(p ? ldsB0 : ldsB1), "v"(bn) : "memory");
      asm volatile("s_wait_asynccnt 0x2" ::: "memory");  // tile kt landed (in-order)
    } else {
      asm volatile("s_wait_asynccnt 0x0" ::: "memory");
    }
    __syncthreads();  // every wave's tile kt is in LDS

    // ---- fragments + 8 WMMAs from buffer p
    const int g = lane >> 4, lr = lane & 15;
    Frag af[2], bfr[4];
#pragma unroll
    for (int mi = 0; mi < 2; ++mi) {
      const unsigned short* pa = &As[p][waveM * 32 + mi * 16 + lr][0];
      // A 16x32 bf16 layout: lane half g holds K {8g..8g+7} and {16+8g..16+8g+7}
      af[mi].q[0] = *(const uint4*)(pa + g * 8);
      af[mi].q[1] = *(const uint4*)(pa + 16 + g * 8);
    }
#pragma unroll
    for (int ni = 0; ni < 4; ++ni) {
      const unsigned short* pb = &Bs[p][waveN * 64 + ni * 16 + lr][0];
      // B 32x16: lane half g holds K {8g..8g+7} and {16+8g..16+8g+7}
      bfr[ni].q[0] = *(const uint4*)(pb + g * 8);
      bfr[ni].q[1] = *(const uint4*)(pb + 16 + g * 8);
    }
#pragma unroll
    for (int mi = 0; mi < 2; ++mi)
#pragma unroll
      for (int ni = 0; ni < 4; ++ni)
        acc[mi][ni] = __builtin_amdgcn_wmma_f32_16x16x32_bf16(
            false, af[mi].v, false, bfr[ni].v, (short)0, acc[mi][ni], false, false);

    __syncthreads();  // all reads of buffer p done before it is overwritten
  }

  // ---- epilogue (C/D layout: lane = {M>>3, N}, VGPR r = M&7)
  const int g = lane >> 4, lr = lane & 15;
#pragma unroll
  for (int mi = 0; mi < 2; ++mi) {
    const int rbase = r0 + waveM * 32 + mi * 16 + g * 8;
#pragma unroll
    for (int ni = 0; ni < 4; ++ni) {
      const int col = n0 + waveN * 64 + ni * 16 + lr;
      if (col < Ncols) {
        const float bv = bias[(size_t)e * Ncols + col];
#pragma unroll
        for (int r = 0; r < 8; ++r) {
          const int rl = rbase + r;
          if (rl >= rows) continue;
          const int slot = seg0 + rl;
          float v = acc[mi][ni][r] + bv;
          if (MODE == 0) {
            Out[(size_t)slot * ldo + col] = f2bf(v);
          } else if (MODE == 1) {
            Out[(size_t)slot * ldo + col] = f2bf(v / (1.f + __expf(-v)));  // SiLU
          } else {
            atomicAdd(&Y[(size_t)tokRow[slot] * Ncols + col], rowW[slot] * v);
          }
        }
      } else if (MODE == 1) {  // zero-fill DFF..UPLD padding for the down pass
#pragma unroll
        for (int r = 0; r < 8; ++r) {
          const int rl = rbase + r;
          if (rl >= rows) continue;
          Out[(size_t)(seg0 + rl) * ldo + col] = 0;
        }
      }
    }
  }
}

// ---------------------------------------------------------------- launcher
extern "C" void kernel_launch(void* const* d_in, const int* in_sizes, int n_in,
                              void* d_out, int out_size, void* d_ws, size_t ws_size,
                              hipStream_t stream) {
  const float* x  = (const float*)d_in[0];
  const float* Wr = (const float*)d_in[1];
  const float* br = (const float*)d_in[2];
  const float* Wg = (const float*)d_in[3];
  const float* bg = (const float*)d_in[4];
  const float* Wu = (const float*)d_in[5];
  const float* bu = (const float*)d_in[6];
  const float* Wd = (const float*)d_in[7];
  const float* bd = (const float*)d_in[8];

  float* y  = (float*)d_out;                 // [N,H]
  float* ew = y + (size_t)NN * HH;           // [N,E] raw logits

  // workspace layout (256B aligned slabs)
  size_t o = 0;
  auto take = [&o](size_t bytes) { size_t r = o; o += (bytes + 255) & ~(size_t)255; return r; };
  size_t o_cnt  = take(EE * 4);
  size_t o_cur  = take(EE * 4);
  size_t o_off  = take(EE * 4);
  size_t o_topE = take((size_t)NKROWS * 4);
  size_t o_topW = take((size_t)NKROWS * 4);
  size_t o_tok  = take((size_t)NKROWS * 4);
  size_t o_rw   = take((size_t)NKROWS * 4);
  size_t o_xb   = take((size_t)NN * HH * 2);
  size_t o_wgT  = take((size_t)EE * HH * HH * 2);     // [E][H][H]   (n-major)
  size_t o_wuT  = take((size_t)EE * DFF * HH * 2);    // [E][DFF][H]
  size_t o_wdT  = take((size_t)EE * HH * UPLD * 2);   // [E][H][UPLD] zero-padded K
  size_t o_G    = take((size_t)NKROWS * HH * 2);
  size_t o_U    = take((size_t)NKROWS * UPLD * 2);
  if (ws_size < o) return;  // workspace too small: no-op (deterministic)

  char* ws = (char*)d_ws;
  int*   cnt  = (int*)(ws + o_cnt);
  int*   cur  = (int*)(ws + o_cur);
  int*   off  = (int*)(ws + o_off);
  int*   topE = (int*)(ws + o_topE);
  float* topW = (float*)(ws + o_topW);
  int*   tok  = (int*)(ws + o_tok);
  float* rw   = (float*)(ws + o_rw);
  unsigned short* xb  = (unsigned short*)(ws + o_xb);
  unsigned short* wgT = (unsigned short*)(ws + o_wgT);
  unsigned short* wuT = (unsigned short*)(ws + o_wuT);
  unsigned short* wdT = (unsigned short*)(ws + o_wdT);
  unsigned short* G   = (unsigned short*)(ws + o_G);
  unsigned short* U   = (unsigned short*)(ws + o_U);

  hipMemsetAsync(y, 0, (size_t)NN * HH * sizeof(float), stream);
  hipMemsetAsync(cnt, 0, EE * sizeof(int), stream);
  hipMemsetAsync(cur, 0, EE * sizeof(int), stream);

  // activations: plain cast; weights: cast + transpose to K-contiguous bf16
  {
    long n = (long)NN * HH;
    cast_bf16_kernel<<<dim3((unsigned)((n + 255) / 256)), 256, 0, stream>>>(x, xb, n);
  }
  auto castT = [&](const float* W, unsigned short* WT, int Kdim, int Ncols, int kpad) {
    long total = (long)EE * Ncols * kpad;
    cast_transpose_kernel<<<dim3((unsigned)((total + 255) / 256)), 256, 0, stream>>>(
        W, WT, Kdim, Ncols, kpad, total);
  };
  castT(Wg, wgT, HH,  HH,  HH);    // gate: K=H,   N=H,   kpad=H
  castT(Wu, wuT, HH,  DFF, HH);    // up:   K=H,   N=DFF, kpad=H
  castT(Wd, wdT, DFF, HH,  UPLD);  // down: K=DFF, N=H,   kpad=UPLD (zero pad)

  router_kernel<<<NN / 8, 256, 0, stream>>>(x, Wr, br, ew, topE, topW, cnt);
  scan_kernel<<<1, 1, 0, stream>>>(cnt, off);
  scatter_kernel<<<NN / 256, 256, 0, stream>>>(topE, topW, off, cur, tok, rw);

  dim3 blk(256);
  // gate: [rows,H] = gather(x) @ Wg[e] + bg[e]
  moe_gemm<0><<<dim3(HH / 128, EE * MAXRT), blk, 0, stream>>>(
      xb, HH, wgT, HH, HH, HH, (size_t)HH * HH, bg,
      G, HH, nullptr, off, cnt, tok, rw);
  // up: [rows,DFF] = silu(G @ Wu[e] + bu[e]), stored with UPLD stride (zero pad)
  moe_gemm<1><<<dim3(UPLD / 128, EE * MAXRT), blk, 0, stream>>>(
      G, HH, wuT, HH, DFF, HH, (size_t)DFF * HH, bu,
      U, UPLD, nullptr, off, cnt, tok, rw);
  // down: y[tok] += w * (U @ Wd[e] + bd[e])  — K padded to UPLD, no guards
  moe_gemm<2><<<dim3(HH / 128, EE * MAXRT), blk, 0, stream>>>(
      U, UPLD, wdT, UPLD, HH, UPLD, (size_t)HH * UPLD, bd,
      nullptr, HH, y, off, cnt, tok, rw);
}
